// Mamba_82523501625676
// MI455X (gfx1250) — matrix operand order repaired and checked by reading, not compiled, not run on hardware
//
#include <hip/hip_runtime.h>

typedef __attribute__((ext_vector_type(16))) __bf16 v16bf;
typedef __attribute__((ext_vector_type(8)))  __bf16 v8bf;
typedef __attribute__((ext_vector_type(8)))  float  v8f;
typedef __attribute__((ext_vector_type(4)))  unsigned v4u;

#define LDT 40  // LDS row stride in bf16 elems (80 B -> 16 B aligned b128 access)

__device__ __forceinline__ unsigned short bfbits(float f) {
  union { float f; unsigned u; } v; v.f = f;
  unsigned r = v.u + 0x7FFFu + ((v.u >> 16) & 1u);   // round-to-nearest-even
  return (unsigned short)(r >> 16);
}
__device__ __forceinline__ unsigned pack2(float a, float b) {
  return (unsigned)bfbits(a) | ((unsigned)bfbits(b) << 16);
}

// C[M,Nd] = A[M,Kd](lda) @ B[Kd,Nd](ldb) + bias, optional softplus, optional +=C
// 256 threads = 8 waves; block tile MT x NT; each wave: MW x NW 16x16 subtiles.
// Double-buffered LDS, software-pipelined (stage k+1 overlaps WMMA on k).
template<bool KG, int MT, int NT, int MW, int NW>
__global__ __launch_bounds__(256)
void gemm_wmma(const float* __restrict__ A, int lda,
               const float* __restrict__ B, int ldb,
               const float* __restrict__ bias,
               float* __restrict__ C, int ldc,
               int Kd, int accum, int act)
{
  constexpr int NWN = (NT / 16) / NW;             // waves along N
  static_assert(((MT / 16) / MW) * NWN == 8, "8 waves");
  __shared__ __bf16 Alds[2][MT * LDT];
  __shared__ __bf16 Blds[2][NT * LDT];

  const int tid  = threadIdx.x;
  const int wave = tid >> 5;
  const int lane = tid & 31;
  const int half = lane >> 4;   // 0: lanes 0-15, 1: lanes 16-31
  const int l16  = lane & 15;
  const int nsb  = (wave % NWN) * NW;
  const int msb  = (wave / NWN) * MW;

  const int m0 = blockIdx.y * MT;
  const int n0 = blockIdx.x * NT;

  // staging coordinates (uniform per thread)
  const int arow = tid >> 2;          // 0..63
  const int akc  = (tid & 3) * 8;     // 0,8,16,24
  const int bn   = tid & 63;
  const int bkb  = (tid >> 6) * 8;    // 0,8,16,24

  // loop-invariant 64-bit offsets (hoisted out of the K loop)
  size_t boff[8];
  #pragma unroll
  for (int j = 0; j < 8; ++j) boff[j] = (size_t)j * ldb;

  // running pointers, advanced by 32 columns / 32 rows per K-step
  const float* Aptr = A + (size_t)(m0 + arow) * lda + akc;
  const float* Bptr = B + (size_t)bkb * ldb + n0 + bn;

  auto stage = [&](int kcur, int p) {
    #pragma unroll
    for (int rr = 0; rr < MT; rr += 64) {          // A tile: MT rows x 32 k
      const float* ap = Aptr + (size_t)rr * lda;
      float v[8];
      if (KG) {
        #pragma unroll
        for (int j = 0; j < 8; ++j) v[j] = (kcur + akc + j < Kd) ? ap[j] : 0.f;
      } else {
        float4 f0 = *(const float4*)ap;
        float4 f1 = *(const float4*)(ap + 4);
        v[0]=f0.x; v[1]=f0.y; v[2]=f0.z; v[3]=f0.w;
        v[4]=f1.x; v[5]=f1.y; v[6]=f1.z; v[7]=f1.w;
      }
      v4u w;
      w[0] = pack2(v[0], v[1]); w[1] = pack2(v[2], v[3]);
      w[2] = pack2(v[4], v[5]); w[3] = pack2(v[6], v[7]);
      *(v4u*)&Alds[p][(rr + arow) * LDT + akc] = w;
    }
    #pragma unroll
    for (int nn = 0; nn < NT; nn += 64) {          // B tile transposed: Blds[n][k]
      const float* bp = Bptr + nn;
      float v[8];
      #pragma unroll
      for (int j = 0; j < 8; ++j) {
        if (KG) v[j] = (kcur + bkb + j < Kd) ? bp[boff[j]] : 0.f;
        else    v[j] = bp[boff[j]];
      }
      v4u w;
      w[0] = pack2(v[0], v[1]); w[1] = pack2(v[2], v[3]);
      w[2] = pack2(v[4], v[5]); w[3] = pack2(v[6], v[7]);
      *(v4u*)&Blds[p][(nn + bn) * LDT + bkb] = w;
    }
    Aptr += 32;
    Bptr += (size_t)32 * ldb;
  };

  v8f acc[MW][NW] = {};

  auto compute = [&](int p) {
    v16bf bf[NW], af[MW];
    #pragma unroll
    for (int nw = 0; nw < NW; ++nw) {
      // B frag (contiguous k, same as A: lanes<16 -> K 0..15 of col l16; lanes>=16 -> K 16..31)
      int n = (nsb + nw) * 16 + l16;
      v8bf lo = *(const v8bf*)&Blds[p][n * LDT + half * 16];
      v8bf hi = *(const v8bf*)&Blds[p][n * LDT + half * 16 + 8];
      #pragma unroll
      for (int j = 0; j < 8; ++j) { bf[nw][j] = lo[j]; bf[nw][j + 8] = hi[j]; }
    }
    #pragma unroll
    for (int mw = 0; mw < MW; ++mw) {
      // A frag (lanes<16 -> K 0..15 of row l16; lanes>=16 -> K 16..31)
      int mrow = (msb + mw) * 16 + l16;
      v8bf lo = *(const v8bf*)&Alds[p][mrow * LDT + half * 16];
      v8bf hi = *(const v8bf*)&Alds[p][mrow * LDT + half * 16 + 8];
      #pragma unroll
      for (int j = 0; j < 8; ++j) { af[mw][j] = lo[j]; af[mw][j + 8] = hi[j]; }
    }
    #pragma unroll
    for (int mw = 0; mw < MW; ++mw)
      #pragma unroll
      for (int nw = 0; nw < NW; ++nw)
        acc[mw][nw] = __builtin_amdgcn_wmma_f32_16x16x32_bf16(
            false, af[mw], false, bf[nw], (short)0, acc[mw][nw], false, false);
  };

  const int nk = (Kd + 31) >> 5;
  stage(0, 0);
  __syncthreads();
  int p = 0;
  for (int it = 0; it < nk - 1; ++it) {            // steady state: unconditional
    stage((it + 1) * 32, p ^ 1);                   // overlaps WMMA below
    compute(p);
    __syncthreads();
    p ^= 1;
  }
  compute(p);                                      // tail

  // ---- epilogue: C/D layout VGPR r -> M = r + 8*half, N = l16 ----
  float bv[NW];
  int   nn[NW];
  #pragma unroll
  for (int nw = 0; nw < NW; ++nw) {
    nn[nw] = n0 + (nsb + nw) * 16 + l16;
    bv[nw] = bias ? bias[nn[nw]] : 0.f;
  }
  float vres[MW][NW][8];
  #pragma unroll
  for (int mw = 0; mw < MW; ++mw)
    #pragma unroll
    for (int nw = 0; nw < NW; ++nw)
      #pragma unroll
      for (int r = 0; r < 8; ++r) vres[mw][nw][r] = acc[mw][nw][r] + bv[nw];
  if (act) {                     // softplus (uniform branch)
    #pragma unroll
    for (int mw = 0; mw < MW; ++mw)
      #pragma unroll
      for (int nw = 0; nw < NW; ++nw)
        #pragma unroll
        for (int r = 0; r < 8; ++r) {
          float v = vres[mw][nw][r];
          vres[mw][nw][r] = (v > 20.f) ? v : __logf(1.f + __expf(v));
        }
  }
  if (accum) {                   // residual accumulate (uniform branch)
    #pragma unroll
    for (int mw = 0; mw < MW; ++mw)
      #pragma unroll
      for (int nw = 0; nw < NW; ++nw)
        #pragma unroll
        for (int r = 0; r < 8; ++r) {
          int m = m0 + (msb + mw) * 16 + r + half * 8;
          vres[mw][nw][r] += C[(size_t)m * ldc + nn[nw]];
        }
  }
  #pragma unroll
  for (int mw = 0; mw < MW; ++mw)
    #pragma unroll
    for (int nw = 0; nw < NW; ++nw)
      #pragma unroll
      for (int r = 0; r < 8; ++r) {
        int m = m0 + (msb + mw) * 16 + r + half * 8;
        C[(size_t)m * ldc + nn[nw]] = vres[mw][nw][r];
      }
}

__global__ void embed_kernel(const int* __restrict__ tok,
                             const float* __restrict__ emb,
                             float* __restrict__ x, int D) {
  int row = blockIdx.x;
  int t = tok[row];
  for (int d = threadIdx.x; d < D; d += blockDim.x)
    x[(size_t)row * D + d] = emb[(size_t)t * D + d];
}

__global__ void rmsnorm_kernel(const float* __restrict__ x,
                               const float* __restrict__ w,
                               float* __restrict__ y, int D) {
  __shared__ float red[256];
  int row = blockIdx.x;
  const float* xr = x + (size_t)row * D;
  float s = 0.f;
  for (int d = threadIdx.x; d < D; d += 256) { float v = xr[d]; s += v * v; }
  red[threadIdx.x] = s;
  __syncthreads();
  for (int off = 128; off > 0; off >>= 1) {
    if (threadIdx.x < off) red[threadIdx.x] += red[threadIdx.x + off];
    __syncthreads();
  }
  float r = rsqrtf(red[0] / (float)D + 1e-6f);
  for (int d = threadIdx.x; d < D; d += 256)
    y[(size_t)row * D + d] = xr[d] * w[d] * r;
}

// u = silu(causal_dwconv(xz[:, :1536]) + cb); xz row stride 3072
__global__ void conv_silu_kernel(const float* __restrict__ xz,
                                 const float* __restrict__ cw,
                                 const float* __restrict__ cb,
                                 float* __restrict__ ua) {
  int idx = blockIdx.x * blockDim.x + threadIdx.x;
  if (idx >= 2048 * 1536) return;
  int l = idx / 1536, e = idx - l * 1536;
  float y = cb[e];
  #pragma unroll
  for (int k = 0; k < 3; ++k) {
    int ls = l + k - 2;
    if (ls >= 0) y += cw[k * 1536 + e] * xz[(size_t)ls * 3072 + e];
  }
  ua[idx] = y / (1.f + __expf(-y));   // silu
}

// ---- chunked selective scan: L=2048 -> 16 chunks x 128 steps ----
// pass A: per (chunk c, channel e): prod(a) and h assuming h_in = 0
__global__ void scan_chunk_kernel(const float* __restrict__ dt,
                                  const float* __restrict__ ua,
                                  const float* __restrict__ dbl,
                                  const float* __restrict__ Alog,
                                  float* __restrict__ pa,
                                  float* __restrict__ hp) {
  int e = blockIdx.x * blockDim.x + threadIdx.x;
  int c = blockIdx.y;
  if (e >= 1536) return;
  float A[8], h[8], p[8];
  #pragma unroll
  for (int n = 0; n < 8; ++n) { A[n] = -__expf(Alog[e * 8 + n]); h[n] = 0.f; p[n] = 1.f; }
  for (int l = c * 128; l < c * 128 + 128; ++l) {
    float dtv = dt[(size_t)l * 1536 + e];
    float dbu = dtv * ua[(size_t)l * 1536 + e];
    #pragma unroll
    for (int n = 0; n < 8; ++n) {
      float a = __expf(dtv * A[n]);
      h[n] = a * h[n] + dbu * dbl[l * 64 + 48 + n];
      p[n] *= a;
    }
  }
  size_t o = ((size_t)c * 1536 + e) * 8;
  #pragma unroll
  for (int n = 0; n < 8; ++n) { pa[o + n] = p[n]; hp[o + n] = h[n]; }
}

// pass B: 16-step sequential prefix over chunks, 12288 independent (e,n) lanes
__global__ void scan_prefix_kernel(const float* __restrict__ pa,
                                   const float* __restrict__ hp,
                                   float* __restrict__ hin) {
  int i = blockIdx.x * blockDim.x + threadIdx.x;
  if (i >= 1536 * 8) return;
  float h = 0.f;
  for (int c = 0; c < 16; ++c) {
    hin[(size_t)c * 1536 * 8 + i] = h;
    h = pa[(size_t)c * 1536 * 8 + i] * h + hp[(size_t)c * 1536 * 8 + i];
  }
}

// pass C: replay chunk with correct h_in; fused u*Dsk skip + silu(res) gate
__global__ void scan_out_kernel(const float* __restrict__ dt,
                                const float* __restrict__ ua,
                                const float* __restrict__ dbl,
                                const float* __restrict__ Alog,
                                const float* __restrict__ Dsk,
                                const float* __restrict__ xz,
                                const float* __restrict__ hin,
                                float* __restrict__ yg) {
  int e = blockIdx.x * blockDim.x + threadIdx.x;
  int c = blockIdx.y;
  if (e >= 1536) return;
  float A[8], h[8];
  size_t o = ((size_t)c * 1536 + e) * 8;
  #pragma unroll
  for (int n = 0; n < 8; ++n) { A[n] = -__expf(Alog[e * 8 + n]); h[n] = hin[o + n]; }
  float dsk = Dsk[e];
  for (int l = c * 128; l < c * 128 + 128; ++l) {
    float dtv = dt[(size_t)l * 1536 + e];
    float uv  = ua[(size_t)l * 1536 + e];
    float dbu = dtv * uv;
    float y = 0.f;
    #pragma unroll
    for (int n = 0; n < 8; ++n) {
      float a = __expf(dtv * A[n]);
      h[n] = a * h[n] + dbu * dbl[l * 64 + 48 + n];
      y += h[n] * dbl[l * 64 + 56 + n];
    }
    y += uv * dsk;
    float r = xz[(size_t)l * 3072 + 1536 + e];
    yg[(size_t)l * 1536 + e] = y * (r / (1.f + __expf(-r)));
  }
}

extern "C" void kernel_launch(void* const* d_in, const int* in_sizes, int n_in,
                              void* d_out, int out_size, void* d_ws, size_t ws_size,
                              hipStream_t stream) {
  const int*   tok     = (const int*)  d_in[0];
  const float* emb     = (const float*)d_in[1];
  const float* W_in    = (const float*)d_in[2];
  const float* b_in    = (const float*)d_in[3];
  const float* conv_w  = (const float*)d_in[4];
  const float* conv_b  = (const float*)d_in[5];
  const float* W_x     = (const float*)d_in[6];
  const float* W_dt    = (const float*)d_in[7];
  const float* b_dt    = (const float*)d_in[8];
  const float* A_log   = (const float*)d_in[9];
  const float* D_skip  = (const float*)d_in[10];
  const float* W_out   = (const float*)d_in[11];
  const float* b_out   = (const float*)d_in[12];
  const float* norm_w  = (const float*)d_in[13];
  const float* normf_w = (const float*)d_in[14];
  const float* W_lm    = (const float*)d_in[15];
  float* out = (float*)d_out;

  float* ws  = (float*)d_ws;
  float* x   = ws; ws += 2048 * 768;    // residual stream
  float* xn  = ws; ws += 2048 * 768;    // rmsnorm output
  float* xz  = ws; ws += 2048 * 3072;   // in-proj (u | res)
  float* ua  = ws; ws += 2048 * 1536;   // silu(conv(u))
  float* dbl = ws; ws += 2048 * 64;     // {dt48, B8, C8}
  float* dtb = ws; ws += 2048 * 1536;   // softplus(dt @ W_dt + b)
  float* yg  = ws; ws += 2048 * 1536;   // gated scan output
  float* pa  = ws; ws += 16 * 1536 * 8; // chunk prod(a)
  float* hp  = ws; ws += 16 * 1536 * 8; // chunk partial h
  float* hin = ws; ws += 16 * 1536 * 8; // chunk entry state

  embed_kernel<<<2048, 256, 0, stream>>>(tok, emb, x, 768);

  for (int i = 0; i < 4; ++i) {
    rmsnorm_kernel<<<2048, 256, 0, stream>>>(x, norm_w + i * 768, xn, 768);

    gemm_wmma<false, 128, 128, 4, 2><<<dim3(3072 / 128, 16), 256, 0, stream>>>(
        xn, 768, W_in + (size_t)i * 768 * 3072, 3072, b_in + i * 3072,
        xz, 3072, 768, 0, 0);

    conv_silu_kernel<<<(2048 * 1536) / 256, 256, 0, stream>>>(
        xz, conv_w + i * 3 * 1536, conv_b + i * 1536, ua);

    gemm_wmma<false, 64, 64, 2, 1><<<dim3(1, 32), 256, 0, stream>>>(
        ua, 1536, W_x + (size_t)i * 1536 * 64, 64, nullptr,
        dbl, 64, 1536, 0, 0);

    // dt = softplus(dbl[:, :48] @ W_dt + b_dt)  (lda=64, K=48 guarded, fused act)
    gemm_wmma<true, 128, 128, 4, 2><<<dim3(1536 / 128, 16), 256, 0, stream>>>(
        dbl, 64, W_dt + (size_t)i * 48 * 1536, 1536, b_dt + i * 1536,
        dtb, 1536, 48, 0, 1);

    scan_chunk_kernel<<<dim3(6, 16), 256, 0, stream>>>(
        dtb, ua, dbl, A_log + (size_t)i * 1536 * 8, pa, hp);
    scan_prefix_kernel<<<48, 256, 0, stream>>>(pa, hp, hin);
    scan_out_kernel<<<dim3(6, 16), 256, 0, stream>>>(
        dtb, ua, dbl, A_log + (size_t)i * 1536 * 8, D_skip + i * 1536, xz, hin, yg);

    // residual: x += yg @ W_out + b_out
    gemm_wmma<false, 128, 128, 4, 2><<<dim3(768 / 128, 16), 256, 0, stream>>>(
        yg, 1536, W_out + (size_t)i * 1536 * 768, 768, b_out + i * 768,
        x, 768, 1536, 1, 0);
  }

  rmsnorm_kernel<<<2048, 256, 0, stream>>>(x, normf_w, xn, 768);

  // LM head: 100 GF, the dominant GEMM
  gemm_wmma<false, 128, 128, 4, 2><<<dim3(32000 / 128, 16), 256, 0, stream>>>(
      xn, 768, W_lm, 32000, nullptr, out, 32000, 768, 0, 0);
}
